// Capsule_26353919328325
// MI455X (gfx1250) — matrix operand order repaired and checked by reading, not compiled, not run on hardware
//
#include <hip/hip_runtime.h>

typedef __attribute__((ext_vector_type(16))) __bf16 v16bf;
typedef __attribute__((ext_vector_type(8)))  float  v8f;
typedef __attribute__((ext_vector_type(4)))  float  v4f;

#define NCAP  5
#define ND    25          // NUM_CAPSULE * DIM_CAPSULE
#define I_DIM 1024
#define K_DIM 128
#define U_STRIDE   28     // pad 25 -> 28 floats: every row 16B aligned in LDS
#define RED_STRIDE 27     // odd-ish stride for reduction scratch
#define SMEM_FLOATS (I_DIM * U_STRIDE + 256 * RED_STRIDE)

__global__ __launch_bounds__(256) void capsule_routing_kernel(
    const float* __restrict__ x,   // (512, 1024, 128)
    const float* __restrict__ W,   // (1, 128, 25)
    float* __restrict__ out)       // (512, 5, 5)
{
    extern __shared__ float smem[];
    float* u_lds = smem;                       // [I_DIM][U_STRIDE]
    float* red   = smem + I_DIM * U_STRIDE;    // [256][RED_STRIDE]

    const int tid  = threadIdx.x;
    const int lane = tid & 31;
    const int wv   = tid >> 5;
    const int hi   = lane >> 4;   // half-wave select
    const int m    = lane & 15;   // row/col within tile
    const int b    = blockIdx.x;

    // ================= Phase 1: u = x[b] @ W via bf16 WMMA =================
    // B fragments: lane l holds B column N = t*16 + l%16, elements K = ks*32 + 8*hi + {0..7, 16..23}.
    v16bf Bf[2][4];
    #pragma unroll
    for (int t = 0; t < 2; ++t) {
        #pragma unroll
        for (int ks = 0; ks < 4; ++ks) {
            const int col = t * 16 + m;
            v16bf f;
            #pragma unroll
            for (int j = 0; j < 16; ++j) {
                const int krow = ks * 32 + 8 * hi + j + (j >= 8 ? 8 : 0);
                const float w = (col < ND) ? W[krow * ND + col] : 0.0f;
                f[j] = (__bf16)w;
            }
            Bf[t][ks] = f;
        }
    }

    const float* xb = x + (size_t)b * (I_DIM * K_DIM);

    #pragma unroll 1
    for (int mt = 0; mt < 8; ++mt) {
        const int Mbase = wv * 128 + mt * 16;
        const int row   = Mbase + m;
        // A-fragment source: lane holds K in [ks*32+8*hi, +8) and [ks*32+16+8*hi, +8)
        const float* rp = xb + row * K_DIM + 8 * hi;
        v8f c0 = {}; v8f c1 = {};
        #pragma unroll
        for (int ks = 0; ks < 4; ++ks) {
            const v4f a0 = *(const v4f*)(rp + ks * 32 +  0);
            const v4f a1 = *(const v4f*)(rp + ks * 32 +  4);
            const v4f a2 = *(const v4f*)(rp + ks * 32 + 16);
            const v4f a3 = *(const v4f*)(rp + ks * 32 + 20);
            v16bf af;
            #pragma unroll
            for (int j = 0; j < 4; ++j) {
                af[j]      = (__bf16)a0[j];
                af[4 + j]  = (__bf16)a1[j];
                af[8 + j]  = (__bf16)a2[j];
                af[12 + j] = (__bf16)a3[j];
            }
            c0 = __builtin_amdgcn_wmma_f32_16x16x32_bf16(false, af, false, Bf[0][ks],
                                                         (short)0, c0, false, false);
            c1 = __builtin_amdgcn_wmma_f32_16x16x32_bf16(false, af, false, Bf[1][ks],
                                                         (short)0, c1, false, false);
        }
        // Scatter C tiles to LDS u: element r <-> (row Mbase + r + 8*hi, col m).
        // c0 columns (0..15) are always valid: branch-free stores.
        #pragma unroll
        for (int r = 0; r < 8; ++r) {
            u_lds[(Mbase + r + 8 * hi) * U_STRIDE + m] = c0[r];
        }
        // c1 columns (16..31): only 16..24 valid. Single hoisted divergent region
        // containing all 8 stores (instead of 8 exec-mask dances).
        if (m < ND - 16) {
            #pragma unroll
            for (int r = 0; r < 8; ++r) {
                u_lds[(Mbase + r + 8 * hi) * U_STRIDE + 16 + m] = c1[r];
            }
        }
    }
    __syncthreads();

    // ================= Phase 2: dynamic routing (4 iterations) =================
    // Thread owns rows i = tid + k*256, k = 0..3. Routing logits in registers.
    float breg[4][NCAP];
    #pragma unroll
    for (int k = 0; k < 4; ++k)
        #pragma unroll
        for (int n = 0; n < NCAP; ++n) breg[k][n] = 0.0f;

    float v[ND];

    #pragma unroll 1
    for (int it = 0; it < 4; ++it) {
        // --- partial s[n,d] = sum_i softmax_n(b)[n,i] * u[i, n*5+d] ---
        float part[ND];
        #pragma unroll
        for (int j = 0; j < ND; ++j) part[j] = 0.0f;

        #pragma unroll 1
        for (int k = 0; k < 4; ++k) {
            const int i = tid + k * 256;
            float mx = breg[k][0];
            #pragma unroll
            for (int n = 1; n < NCAP; ++n) mx = fmaxf(mx, breg[k][n]);
            float e[NCAP], es = 0.0f;
            #pragma unroll
            for (int n = 0; n < NCAP; ++n) { e[n] = __expf(breg[k][n] - mx); es += e[n]; }
            const float inv = 1.0f / es;

            const float* up = u_lds + i * U_STRIDE;   // 112B-aligned
            float uv[ND];
            #pragma unroll
            for (int q = 0; q < 6; ++q) {
                const v4f u4 = *(const v4f*)(up + 4 * q);
                #pragma unroll
                for (int j = 0; j < 4; ++j) uv[4 * q + j] = u4[j];
            }
            uv[24] = up[24];

            #pragma unroll
            for (int n = 0; n < NCAP; ++n) {
                const float cn = e[n] * inv;
                #pragma unroll
                for (int d = 0; d < NCAP; ++d)
                    part[n * 5 + d] = fmaf(cn, uv[n * 5 + d], part[n * 5 + d]);
            }
        }

        // --- block tree reduction of part[25] ---
        #pragma unroll
        for (int j = 0; j < ND; ++j) red[tid * RED_STRIDE + j] = part[j];
        __syncthreads();
        for (int st = 128; st > 0; st >>= 1) {
            if (tid < st) {
                #pragma unroll
                for (int j = 0; j < ND; ++j)
                    red[tid * RED_STRIDE + j] += red[(tid + st) * RED_STRIDE + j];
            }
            __syncthreads();
        }

        // --- squash (redundantly in every thread): v = s / sqrt(|s|^2 + eps) ---
        #pragma unroll
        for (int n = 0; n < NCAP; ++n) {
            float norm = 1e-7f;
            #pragma unroll
            for (int d = 0; d < NCAP; ++d) {
                const float sv = red[n * 5 + d];
                norm = fmaf(sv, sv, norm);
            }
            const float rn = 1.0f / sqrtf(norm);
            #pragma unroll
            for (int d = 0; d < NCAP; ++d) v[n * 5 + d] = red[n * 5 + d] * rn;
        }

        if (it == 3) {
            if (tid < ND) out[b * ND + tid] = v[tid];
        } else {
            // --- b[n,i] = sum_d v[n,d] * u[i, n*5+d] ---
            #pragma unroll 1
            for (int k = 0; k < 4; ++k) {
                const int i = tid + k * 256;
                const float* up = u_lds + i * U_STRIDE;
                float uv[ND];
                #pragma unroll
                for (int q = 0; q < 6; ++q) {
                    const v4f u4 = *(const v4f*)(up + 4 * q);
                    #pragma unroll
                    for (int j = 0; j < 4; ++j) uv[4 * q + j] = u4[j];
                }
                uv[24] = up[24];
                #pragma unroll
                for (int n = 0; n < NCAP; ++n) {
                    float acc = 0.0f;
                    #pragma unroll
                    for (int d = 0; d < NCAP; ++d)
                        acc = fmaf(v[n * 5 + d], uv[n * 5 + d], acc);
                    breg[k][n] = acc;
                }
            }
        }
        __syncthreads();   // protect red[] reads before next iteration rewrites it
    }
}

extern "C" void kernel_launch(void* const* d_in, const int* in_sizes, int n_in,
                              void* d_out, int out_size, void* d_ws, size_t ws_size,
                              hipStream_t stream) {
    const float* x = (const float*)d_in[0];   // (512, 1024, 128) f32
    const float* W = (const float*)d_in[1];   // (1, 128, 25)     f32
    float* out = (float*)d_out;               // (512, 5, 5)      f32

    const int B = in_sizes[0] / (I_DIM * K_DIM);   // 512
    const size_t smem_bytes = (size_t)SMEM_FLOATS * sizeof(float);  // ~139 KB < 320 KB/WGP

    capsule_routing_kernel<<<dim3(B), dim3(256), smem_bytes, stream>>>(x, W, out);
}
